// AttentionFusion_3822520893965
// MI455X (gfx1250) — matrix operand order repaired and checked by reading, not compiled, not run on hardware
//
#include <hip/hip_runtime.h>
#include <hip/hip_bf16.h>

typedef __bf16 bf16;
typedef __attribute__((ext_vector_type(16))) __bf16 v16bf;
typedef __attribute__((ext_vector_type(8)))  __bf16 bf16x8;
typedef __attribute__((ext_vector_type(4)))  __bf16 bf16x4;
typedef __attribute__((ext_vector_type(8)))  float  v8f;

// Problem constants
#define CDIM 1024
#define BDIM 32
#define HWDIM 1024

// Async global->LDS copy of 16B per lane (CDNA5, ASYNCcnt-tracked).
// ldsoff = per-lane LDS byte offset (VGPR); gaddr = 64-bit global address.
#define ASYNC_CP_B128(gaddr, ldsoff)                                       \
  asm volatile("global_load_async_to_lds_b128 %0, %1, off"                 \
               :: "v"(ldsoff), "v"(gaddr) : "memory")

#define WAIT_ASYNC0() asm volatile("s_wait_asynccnt 0x0" ::: "memory")

// ---------------------------------------------------------------------------
// f32 -> bf16 elementwise convert (weights)
// ---------------------------------------------------------------------------
__global__ __launch_bounds__(256) void convert_f32_bf16(
    const float* __restrict__ in, bf16* __restrict__ out, int n) {
  int i = (blockIdx.x * 256 + threadIdx.x) * 4;
  if (i + 3 < n) {
    const float4 v = *reinterpret_cast<const float4*>(in + i);
    bf16x4 o;
    o[0] = (bf16)v.x; o[1] = (bf16)v.y; o[2] = (bf16)v.z; o[3] = (bf16)v.w;
    *reinterpret_cast<bf16x4*>(out + i) = o;
  }
}

// ---------------------------------------------------------------------------
// f32 [b][c][s] -> bf16 [b][s][c]  (transpose-convert, 32x32 LDS tiles)
// ---------------------------------------------------------------------------
__global__ __launch_bounds__(256) void transpose_convert(
    const float* __restrict__ in, bf16* __restrict__ outT) {
  __shared__ float tile[32][33];
  const size_t b = blockIdx.z;
  const float* inb = in + b * (size_t)(CDIM * HWDIM);
  bf16* outb = outT + b * (size_t)(CDIM * HWDIM);
  const int s0 = blockIdx.x * 32, c0 = blockIdx.y * 32;
  const int tx = threadIdx.x, ty = threadIdx.y;
#pragma unroll
  for (int j = 0; j < 32; j += 8)
    tile[ty + j][tx] = inb[(size_t)(c0 + ty + j) * HWDIM + s0 + tx];
  __syncthreads();
#pragma unroll
  for (int j = 0; j < 32; j += 8)
    outb[(size_t)(s0 + ty + j) * CDIM + c0 + tx] = (bf16)tile[tx][ty + j];
}

// ---------------------------------------------------------------------------
// Batched TN GEMM:  Out[b][M][N] = A[b][M][K] * B'[b][N][K]^T   (M=N=K=1024)
// bf16 inputs, f32 WMMA accumulation, epilogue variants.
// 128x128 tile per workgroup, 8 waves (wave32), 64x32 per wave, BK=32.
// Tiles streamed with GLOBAL_LOAD_ASYNC_TO_LDS_B128 into an LDS double buffer
// (one barrier per K-step); global addresses advance by pointer increment so
// no WMMA-source VGPRs are rewritten (avoids WMMA->VALU WAR hazard NOPs).
// ---------------------------------------------------------------------------
enum { EPI_TANH = 0, EPI_TANH_T = 1, EPI_SCALE_F32 = 2, EPI_T = 3, EPI_BIAS_RES = 4 };

template <int EPI>
__global__ __launch_bounds__(256, 2) void gemm_tn(
    const bf16* __restrict__ A, long long sAb,
    const bf16* __restrict__ Bt, long long sBb,
    const float* __restrict__ bias,
    const float* __restrict__ res, long long sRb,
    void* __restrict__ OutP, long long sOb,
    float scale) {
  constexpr int LDSS = 40;  // bf16 elems per LDS row: 32 + 8 pad (80B pitch)
  __shared__ __align__(16) bf16 lA[2][128 * LDSS];
  __shared__ __align__(16) bf16 lB[2][128 * LDSS];

  const int b = blockIdx.z;
  const int tileN = blockIdx.x * 128;
  const int tileM = blockIdx.y * 128;
  const int t = threadIdx.x;
  const int lane = t & 31;
  const int wave = t >> 5;
  const int laneLo = lane & 15, laneHi = lane >> 4;
  const int wm = (wave >> 2) * 64;  // 0 or 64
  const int wn = (wave & 3) * 32;   // 0,32,64,96

  const bf16* Ab = A + (size_t)b * sAb;
  const bf16* Bb = Bt + (size_t)b * sBb;

  // Each thread copies 2x16B of the A-tile and 2x16B of the B-tile per K-step.
  const int r0 = t >> 2;         // 0..63
  const int c0 = (t & 3) << 3;   // 0,8,16,24

  // Persistent per-thread copy addresses (advance by 64B per issued K-slice).
  unsigned long long gA0 = (unsigned long long)(Ab + (size_t)(tileM + r0) * CDIM + c0);
  unsigned long long gA1 = gA0 + (unsigned long long)64 * CDIM * sizeof(bf16);
  unsigned long long gB0 = (unsigned long long)(Bb + (size_t)(tileN + r0) * CDIM + c0);
  unsigned long long gB1 = gB0 + (unsigned long long)64 * CDIM * sizeof(bf16);
  const unsigned dA0[2] = {
      (unsigned)(unsigned long long)&lA[0][r0 * LDSS + c0],
      (unsigned)(unsigned long long)&lA[1][r0 * LDSS + c0]};
  const unsigned dA1[2] = {dA0[0] + 64u * LDSS * 2u, dA0[1] + 64u * LDSS * 2u};
  const unsigned dB0[2] = {
      (unsigned)(unsigned long long)&lB[0][r0 * LDSS + c0],
      (unsigned)(unsigned long long)&lB[1][r0 * LDSS + c0]};
  const unsigned dB1[2] = {dB0[0] + 64u * LDSS * 2u, dB0[1] + 64u * LDSS * 2u};

  v8f acc[4][2];
#pragma unroll
  for (int mf = 0; mf < 4; ++mf)
#pragma unroll
    for (int nf = 0; nf < 2; ++nf)
#pragma unroll
      for (int i = 0; i < 8; ++i) acc[mf][nf][i] = 0.0f;

  auto issueStep = [&](int buf) {
    ASYNC_CP_B128(gA0, dA0[buf]);
    ASYNC_CP_B128(gA1, dA1[buf]);
    ASYNC_CP_B128(gB0, dB0[buf]);
    ASYNC_CP_B128(gB1, dB1[buf]);
    gA0 += 64ull; gA1 += 64ull; gB0 += 64ull; gB1 += 64ull;
  };

  auto computeTile = [&](int buf) {
    // A fragment: lane<16 row M=laneLo holds K{0..7,16..23}; lane>=16 K{8..15,24..31}
    v16bf af[4];
#pragma unroll
    for (int mf = 0; mf < 4; ++mf) {
      const bf16* base = &lA[buf][(wm + mf * 16 + laneLo) * LDSS];
      bf16x8 lo = *reinterpret_cast<const bf16x8*>(base + laneHi * 8);
      bf16x8 hi = *reinterpret_cast<const bf16x8*>(base + 16 + laneHi * 8);
#pragma unroll
      for (int i = 0; i < 8; ++i) { af[mf][i] = lo[i]; af[mf][i + 8] = hi[i]; }
    }
    // B fragment: lane<16 col N=laneLo holds K{0..7,16..23}; lane>=16 K{8..15,24..31}
    v16bf bfr[2];
#pragma unroll
    for (int nf = 0; nf < 2; ++nf) {
      const bf16* base = &lB[buf][(wn + nf * 16 + laneLo) * LDSS + laneHi * 8];
      bf16x8 lo = *reinterpret_cast<const bf16x8*>(base);
      bf16x8 hi = *reinterpret_cast<const bf16x8*>(base + 16);
#pragma unroll
      for (int i = 0; i < 8; ++i) { bfr[nf][i] = lo[i]; bfr[nf][i + 8] = hi[i]; }
    }
#pragma unroll
    for (int mf = 0; mf < 4; ++mf)
#pragma unroll
      for (int nf = 0; nf < 2; ++nf)
        acc[mf][nf] = __builtin_amdgcn_wmma_f32_16x16x32_bf16(
            false, af[mf], false, bfr[nf], (short)0, acc[mf][nf], false, false);
  };

  issueStep(0);  // tile k=0 -> buf0
#pragma unroll 1
  for (int k0 = 0; k0 < CDIM; k0 += 64) {
    WAIT_ASYNC0();            // our async writes into buf0 have landed
    __syncthreads();          // everyone's have landed; prior buf0 reads retired
    issueStep(1);             // prefetch k0+32 into buf1
    computeTile(0);
    WAIT_ASYNC0();
    __syncthreads();
    if (k0 + 64 < CDIM) issueStep(0);  // prefetch k0+64 into buf0
    computeTile(1);
  }

  // Epilogue.  C/D layout: VGPR i, lanes 0-15: (M=i, N=lane); lanes 16-31: (M=8+i).
#pragma unroll
  for (int mf = 0; mf < 4; ++mf) {
#pragma unroll
    for (int nf = 0; nf < 2; ++nf) {
      const int n = tileN + wn + nf * 16 + laneLo;
      const int mBase = tileM + wm + mf * 16 + laneHi * 8;
      if constexpr (EPI == EPI_TANH) {
        bf16* out = (bf16*)OutP + (size_t)b * sOb;
#pragma unroll
        for (int i = 0; i < 8; ++i) {
          const int m = mBase + i;
          out[(size_t)m * CDIM + n] = (bf16)tanhf(acc[mf][nf][i] + bias[m]);
        }
      } else if constexpr (EPI == EPI_TANH_T) {
        bf16* out = (bf16*)OutP + (size_t)b * sOb;
        bf16x8 pk;
#pragma unroll
        for (int i = 0; i < 8; ++i)
          pk[i] = (bf16)tanhf(acc[mf][nf][i] + bias[mBase + i]);
        *reinterpret_cast<bf16x8*>(out + (size_t)n * CDIM + mBase) = pk;
      } else if constexpr (EPI == EPI_SCALE_F32) {
        float* out = (float*)OutP + (size_t)b * sOb;
#pragma unroll
        for (int i = 0; i < 8; ++i)
          out[(size_t)(mBase + i) * CDIM + n] = acc[mf][nf][i] * scale;
      } else if constexpr (EPI == EPI_T) {
        bf16* out = (bf16*)OutP + (size_t)b * sOb;
        bf16x8 pk;
#pragma unroll
        for (int i = 0; i < 8; ++i) pk[i] = (bf16)acc[mf][nf][i];
        *reinterpret_cast<bf16x8*>(out + (size_t)n * CDIM + mBase) = pk;
      } else {  // EPI_BIAS_RES
        float* out = (float*)OutP + (size_t)b * sOb;
        const float* rsb = res + (size_t)b * sRb;
#pragma unroll
        for (int i = 0; i < 8; ++i) {
          const int m = mBase + i;
          out[(size_t)m * CDIM + n] =
              acc[mf][nf][i] + bias[m] + rsb[(size_t)m * CDIM + n];
        }
      }
    }
  }
  (void)scale;
}

// ---------------------------------------------------------------------------
// Row softmax: one 256-thread block per 1024-wide f32 row -> bf16 row
// ---------------------------------------------------------------------------
__global__ __launch_bounds__(256) void softmax_rows(
    const float* __restrict__ S, bf16* __restrict__ P) {
  __shared__ float red[8];
  const size_t row = blockIdx.x;
  const float* x = S + row * CDIM;
  const int t = threadIdx.x;
  float v[4];
  float mx = -3.402823466e38f;
#pragma unroll
  for (int i = 0; i < 4; ++i) { v[i] = x[t + i * 256]; mx = fmaxf(mx, v[i]); }
#pragma unroll
  for (int o = 16; o; o >>= 1) mx = fmaxf(mx, __shfl_xor(mx, o, 32));
  if ((t & 31) == 0) red[t >> 5] = mx;
  __syncthreads();
  mx = red[0];
#pragma unroll
  for (int i = 1; i < 8; ++i) mx = fmaxf(mx, red[i]);
  float sum = 0.0f;
#pragma unroll
  for (int i = 0; i < 4; ++i) { v[i] = __expf(v[i] - mx); sum += v[i]; }
#pragma unroll
  for (int o = 16; o; o >>= 1) sum += __shfl_xor(sum, o, 32);
  __syncthreads();
  if ((t & 31) == 0) red[t >> 5] = sum;
  __syncthreads();
  sum = 0.0f;
#pragma unroll
  for (int i = 0; i < 8; ++i) sum += red[i];
  const float inv = 1.0f / sum;
  bf16* p = P + row * CDIM;
#pragma unroll
  for (int i = 0; i < 4; ++i) p[t + i * 256] = (bf16)(v[i] * inv);
}

// ---------------------------------------------------------------------------
// Launch
// ---------------------------------------------------------------------------
extern "C" void kernel_launch(void* const* d_in, const int* in_sizes, int n_in,
                              void* d_out, int out_size, void* d_ws, size_t ws_size,
                              hipStream_t stream) {
  (void)in_sizes; (void)n_in; (void)out_size; (void)ws_size;
  const float* shape = (const float*)d_in[0];
  const float* img   = (const float*)d_in[1];
  const float* wq = (const float*)d_in[2];
  const float* bq = (const float*)d_in[3];
  const float* wk = (const float*)d_in[4];
  const float* bk = (const float*)d_in[5];
  const float* wv = (const float*)d_in[6];
  const float* bv = (const float*)d_in[7];
  const float* wc = (const float*)d_in[8];
  const float* bc = (const float*)d_in[9];
  float* out = (float*)d_out;

  constexpr size_t MB = 1024ull * 1024ull;
  char* ws = (char*)d_ws;
  bf16* Wq  = (bf16*)(ws + 0 * MB);
  bf16* Wk  = (bf16*)(ws + 2 * MB);
  bf16* Wv  = (bf16*)(ws + 4 * MB);
  bf16* Wc  = (bf16*)(ws + 6 * MB);
  bf16* XsT = (bf16*)(ws + 8 * MB);     // [b][s][c] bf16, 64 MB
  bf16* XiT = (bf16*)(ws + 72 * MB);    // [b][s][c] bf16, 64 MB
  bf16* Q   = (bf16*)(ws + 136 * MB);   // [b][c][s] bf16, 64 MB (reused for NVT)
  bf16* K   = (bf16*)(ws + 200 * MB);   // [b][c][s] bf16, 64 MB (reused for P)
  bf16* VT  = (bf16*)(ws + 264 * MB);   // [b][s][c] bf16, 64 MB
  float* S  = (float*)(ws + 328 * MB);  // [b][c][c] f32, 128 MB
  bf16* P   = K;                        // softmax output (K is dead)
  bf16* NVT = Q;                        // P*V output, transposed (Q is dead)

  const long long M1 = (long long)(CDIM * CDIM);  // 1M elements per batch matrix

  // Stage 0: weight & input conversions
  convert_f32_bf16<<<1024, 256, 0, stream>>>(wq, Wq, CDIM * CDIM);
  convert_f32_bf16<<<1024, 256, 0, stream>>>(wk, Wk, CDIM * CDIM);
  convert_f32_bf16<<<1024, 256, 0, stream>>>(wv, Wv, CDIM * CDIM);
  convert_f32_bf16<<<1024, 256, 0, stream>>>(wc, Wc, CDIM * CDIM);
  transpose_convert<<<dim3(32, 32, BDIM), dim3(32, 8), 0, stream>>>(shape, XsT);
  transpose_convert<<<dim3(32, 32, BDIM), dim3(32, 8), 0, stream>>>(img, XiT);

  const dim3 gg(8, 8, BDIM), gb(256);
  // Stage 1: QKV convs (W[o,c] x X^T[s,c]^T), tanh(+bias)
  gemm_tn<EPI_TANH><<<gg, gb, 0, stream>>>(Wq, 0, XsT, M1, bq, nullptr, 0, Q, M1, 0.f);
  gemm_tn<EPI_TANH><<<gg, gb, 0, stream>>>(Wk, 0, XiT, M1, bk, nullptr, 0, K, M1, 0.f);
  gemm_tn<EPI_TANH_T><<<gg, gb, 0, stream>>>(Wv, 0, XiT, M1, bv, nullptr, 0, VT, M1, 0.f);
  // Stage 2: scores = Q[c,s] x K[d,s]^T / sqrt(C)
  gemm_tn<EPI_SCALE_F32><<<gg, gb, 0, stream>>>(Q, M1, K, M1, nullptr, nullptr, 0, S, M1, 0.03125f);
  // Stage 3: softmax over d
  softmax_rows<<<BDIM * CDIM, 256, 0, stream>>>(S, P);
  // Stage 4: newV^T = (P[c,d] x V^T[s,d]^T)^T
  gemm_tn<EPI_T><<<gg, gb, 0, stream>>>(P, M1, VT, M1, nullptr, nullptr, 0, NVT, M1, 0.f);
  // Stage 5: out = Wc x NV + bc + shape_map
  gemm_tn<EPI_BIAS_RES><<<gg, gb, 0, stream>>>(Wc, 0, NVT, M1, bc, shape, M1, out, M1, 0.f);
}
